// CrossAttentionBlock_54030688584321
// MI455X (gfx1250) — matrix operand order repaired and checked by reading, not compiled, not run on hardware
//
#include <hip/hip_runtime.h>
#include <hip/hip_bf16.h>

// ---------------------------------------------------------------------------
// CDNA5 (gfx1250) cross-attention block.
// - All matmuls: v_wmma_f32_16x16x32_bf16 (wave32 WMMA), fp32 accumulate.
// - Intermediates held in bf16 (halves HBM traffic vs fp32; math is bf16
//   anyway). fp32 inputs/weights are cast (and weights transposed) once.
// - bf16 operand staging uses GLOBAL_LOAD_ASYNC_TO_LDS_B128 (ASYNCcnt),
//   double-buffered LDS, one barrier per K-step; WMMA overlaps the copies.
// ---------------------------------------------------------------------------

typedef __attribute__((ext_vector_type(8)))  float  v8f;
typedef __attribute__((ext_vector_type(4)))  __bf16 v4bf;
typedef __attribute__((ext_vector_type(8)))  __bf16 v8bf;
typedef __attribute__((ext_vector_type(16))) __bf16 v16bf;

#define TILE_M 128
#define TILE_N 128
#define TILE_K 32
#define LDSP   40   // bf16 elems per LDS row (80 B): 16B-aligned, conflict-free

__device__ __forceinline__ __bf16 f2bf1(float f) {
    union { float f; unsigned u; } c; c.f = f;
    unsigned r = c.u + 0x7fffu + ((c.u >> 16) & 1u);   // round-to-nearest-even
    unsigned short h = (unsigned short)(r >> 16);
    __bf16 b;
    __builtin_memcpy(&b, &h, 2);
    return b;
}

__device__ __forceinline__ void cvt4store(__bf16* d, float4 v) {
    d[0] = f2bf1(v.x); d[1] = f2bf1(v.y); d[2] = f2bf1(v.z); d[3] = f2bf1(v.w);
}

// Async global->LDS copy of 16 bytes (8 bf16), tracked by ASYNCcnt.
__device__ __forceinline__ void async_cp16(const __bf16* l, const __bf16* g) {
    auto p3 = (const __attribute__((address_space(3))) __bf16*)l;   // LDS offset (32-bit)
    asm volatile("global_load_async_to_lds_b128 %0, %1, off"
                 :: "v"(p3), "v"(g)
                 : "memory");
}
__device__ __forceinline__ void wait_async0() {
    asm volatile("s_wait_asynccnt 0" ::: "memory");
}

// Generic tiled GEMM:  C = epilogue( scale * A[M,K] x B + bias )
//   AMODE 1: A bf16 row-major [M,K], async copy-stage.
//   AMODE 0: A fp32 row-major [M,K], VGPR convert-stage (attn probabilities).
//   BMODE 0: B bf16 memory [N,K] row-major, async copy-stage.
//   BMODE 1: B bf16 memory [K,N] row-major, register transpose-stage (V).
//   CMODE 0: C fp32; CMODE 1: C bf16.
//   Batched over blockIdx.z with (b = z/H, h = z%H) element strides.
//   Epilogue: *scale, +bias[n], key-mask -> -1e30, LeakyReLU, +residual(fp32).
template <int AMODE, int BMODE, int CMODE>
__global__ __launch_bounds__(256)
void gemm_wmma_kernel(
    const void*   __restrict__ Av,   int lda, long long sAb, long long sAh,
    const __bf16* __restrict__ Bmat, int ldb, long long sBb, long long sBh,
    void*         __restrict__ Cv,   int ldc, long long sCb, long long sCh,
    const float*  __restrict__ bias,
    const float*  __restrict__ residual, int ldr,
    const int*    __restrict__ maskN, int maskStride,
    int K, int H, float scale, int applyLeaky)
{
    constexpr bool ANY_ASYNC = (AMODE == 1) || (BMODE == 0);

    __shared__ __align__(16) __bf16 At[2][TILE_M * LDSP];
    __shared__ __align__(16) __bf16 Bt[2][TILE_N * LDSP];

    const int tid   = threadIdx.x;
    const int lane  = tid & 31;
    const int wid   = tid >> 5;
    const int wm    = wid >> 1;        // 0..3  -> 32-row strip
    const int wn    = wid & 1;         // 0..1  -> 64-col strip
    const int lhalf = lane >> 4;       // 0/1
    const int l16   = lane & 15;

    const long long tileN = (long long)blockIdx.x * TILE_N;
    const long long tileM = (long long)blockIdx.y * TILE_M;
    const int z    = blockIdx.z;
    const int bidx = z / H;
    const int hidx = z % H;

    const __bf16* Bb = Bmat + (long long)bidx * sBb + (long long)hidx * sBh;

    // staging geometry
    const int rowS = tid >> 3, colS = (tid & 7) << 2;    // fp32-A: float4 slots
    const int rowY = tid >> 2, colY = (tid & 3) << 3;    // async: 16B chunks (4/row)
    const int kTb  = tid >> 4, cTb  = (tid & 15) << 3;   // transpose-B: 8 bf16 along n

    const float*  aF = nullptr;
    const __bf16* aY = nullptr;
    const __bf16* bY = nullptr;
    const __bf16* bT = nullptr;
    if constexpr (AMODE == 0) {
        const float* Afp = (const float*)Av + (long long)bidx * sAb + (long long)hidx * sAh;
        aF = Afp + (tileM + rowS) * (long long)lda + colS;
    } else {
        const __bf16* Abf = (const __bf16*)Av + (long long)bidx * sAb + (long long)hidx * sAh;
        aY = Abf + (tileM + rowY) * (long long)lda + colY;
    }
    if constexpr (BMODE == 0) {
        bY = Bb + (tileN + rowY) * (long long)ldb + colY;
    } else {
        bT = Bb + (long long)kTb * ldb + tileN + cTb;
    }

    v8f zero;
#pragma unroll
    for (int e = 0; e < 8; ++e) zero[e] = 0.0f;
    v8f acc[2][4];
#pragma unroll
    for (int mi = 0; mi < 2; ++mi)
#pragma unroll
        for (int ni = 0; ni < 4; ++ni) acc[mi][ni] = zero;

    float4 aReg[4];
    v8bf   bReg[2];

    auto stageIssue = [&](int buf, int k0) {
        if constexpr (AMODE == 1) {
#pragma unroll
            for (int i = 0; i < 2; ++i)
                async_cp16(&At[buf][(rowY + 64 * i) * LDSP + colY],
                           aY + (long long)(64 * i) * lda + k0);
        } else {
#pragma unroll
            for (int i = 0; i < 4; ++i)
                aReg[i] = *(const float4*)(aF + (long long)(32 * i) * lda + k0);
        }
        if constexpr (BMODE == 0) {
#pragma unroll
            for (int i = 0; i < 2; ++i)
                async_cp16(&Bt[buf][(rowY + 64 * i) * LDSP + colY],
                           bY + (long long)(64 * i) * ldb + k0);
        } else {
#pragma unroll
            for (int i = 0; i < 2; ++i)
                bReg[i] = *(const v8bf*)(bT + (long long)(k0 + 16 * i) * ldb);
        }
    };

    auto stageFinish = [&](int buf) {
        if constexpr (AMODE == 0) {
#pragma unroll
            for (int i = 0; i < 4; ++i)
                cvt4store(&At[buf][(rowS + 32 * i) * LDSP + colS], aReg[i]);
        }
        if constexpr (BMODE == 1) {
#pragma unroll
            for (int i = 0; i < 2; ++i) {
                const int kk = kTb + 16 * i;
#pragma unroll
                for (int j = 0; j < 8; ++j)
                    Bt[buf][(cTb + j) * LDSP + kk] = bReg[i][j];
            }
        }
    };

    auto compute = [&](int buf) {
        // fragments per ISA layout: two 16B runs for A, two 16B runs for B
        v16bf afrag[2], bfrag[4];
#pragma unroll
        for (int mi = 0; mi < 2; ++mi) {
            const __bf16* p = &At[buf][(wm * 32 + mi * 16 + l16) * LDSP + lhalf * 8];
            v8bf lo = *(const v8bf*)p;                  // K = half*8 .. +7
            v8bf hi = *(const v8bf*)(p + 16);           // K = 16+half*8 .. +7
            afrag[mi] = __builtin_shufflevector(lo, hi, 0,1,2,3,4,5,6,7,8,9,10,11,12,13,14,15);
        }
#pragma unroll
        for (int ni = 0; ni < 4; ++ni) {
            const __bf16* p = &Bt[buf][(wn * 64 + ni * 16 + l16) * LDSP + lhalf * 8];
            v8bf lo = *(const v8bf*)p;                  // K = half*8 .. +7
            v8bf hi = *(const v8bf*)(p + 16);           // K = 16+half*8 .. +7
            bfrag[ni] = __builtin_shufflevector(lo, hi, 0,1,2,3,4,5,6,7,8,9,10,11,12,13,14,15);
        }
#pragma unroll
        for (int mi = 0; mi < 2; ++mi)
#pragma unroll
            for (int ni = 0; ni < 4; ++ni)
                acc[mi][ni] = __builtin_amdgcn_wmma_f32_16x16x32_bf16(
                    false, afrag[mi], false, bfrag[ni], (short)0, acc[mi][ni], false, false);
    };

    // ---- software-pipelined main loop (double-buffered LDS) ----
    stageIssue(0, 0);
    stageFinish(0);
    if constexpr (ANY_ASYNC) wait_async0();
    __syncthreads();
    int cur = 0;
    for (int k0 = TILE_K; k0 < K; k0 += TILE_K) {
        stageIssue(cur ^ 1, k0);    // async copies / global loads fly ...
        compute(cur);               // ... under the WMMAs of tile k
        stageFinish(cur ^ 1);
        if constexpr (ANY_ASYNC) wait_async0();
        __syncthreads();
        cur ^= 1;
    }
    compute(cur);

    // ---- fused epilogue ----
#pragma unroll
    for (int mi = 0; mi < 2; ++mi) {
#pragma unroll
        for (int ni = 0; ni < 4; ++ni) {
            const long long colL    = tileN + wn * 64 + ni * 16 + l16;
            const long long rowBase = tileM + wm * 32 + mi * 16 + lhalf * 8;
            const float addb = bias ? bias[colL] : 0.0f;
            const int mskBad = maskN ? (maskN[(long long)bidx * maskStride + colL] == 0) : 0;
#pragma unroll
            for (int e = 0; e < 8; ++e) {
                const long long row = rowBase + e;
                float v = acc[mi][ni][e] * scale + addb;
                if (mskBad) v = -1.0e30f;
                if (applyLeaky) v = (v > 0.0f) ? v : v * 0.01f;
                if (residual) v += residual[row * (long long)ldr + colL];
                if constexpr (CMODE == 0) {
                    float* Cb = (float*)Cv + (long long)bidx * sCb + (long long)hidx * sCh;
                    Cb[row * (long long)ldc + colL] = v;
                } else {
                    __bf16* Cb = (__bf16*)Cv + (long long)bidx * sCb + (long long)hidx * sCh;
                    Cb[row * (long long)ldc + colL] = f2bf1(v);
                }
            }
        }
    }
}

// fp32 -> bf16 elementwise cast (activations). n4 = count of float4 groups.
__global__ __launch_bounds__(256)
void cast_f32_bf16_kernel(const float* __restrict__ src, __bf16* __restrict__ dst,
                          long long n4)
{
    const long long i = (long long)blockIdx.x * 256 + threadIdx.x;
    if (i >= n4) return;
    const float4 v = ((const float4*)src)[i];
    cvt4store(dst + i * 4, v);
}

// fp32 [K,N]=1024x1024 weight -> bf16 [N,K] transpose (LDS 32x33 tile).
__global__ __launch_bounds__(256)
void transpose_cast_kernel(const float* __restrict__ W, __bf16* __restrict__ Wt)
{
    __shared__ float t[32][33];
    const int r  = threadIdx.x >> 3;
    const int c4 = (threadIdx.x & 7) << 2;
    const long long kBase = (long long)blockIdx.y * 32;
    const long long nBase = (long long)blockIdx.x * 32;
    const float4 v = *(const float4*)(W + (kBase + r) * 1024 + nBase + c4);
    t[r][c4 + 0] = v.x; t[r][c4 + 1] = v.y; t[r][c4 + 2] = v.z; t[r][c4 + 3] = v.w;
    __syncthreads();
    __bf16* d = Wt + (nBase + r) * 1024 + kBase + c4;
    d[0] = f2bf1(t[c4 + 0][r]); d[1] = f2bf1(t[c4 + 1][r]);
    d[2] = f2bf1(t[c4 + 2][r]); d[3] = f2bf1(t[c4 + 3][r]);
}

// In-place row softmax (fp32, in d_out), one wave32 per row.
template <int LK>
__global__ __launch_bounds__(256)
void softmax_rows_kernel(float* __restrict__ data, long long nRows)
{
    const int lane = threadIdx.x & 31;
    const int wid  = threadIdx.x >> 5;
    const long long row = (long long)blockIdx.x * 8 + wid;
    if (row >= nRows) return;
    float* p = data + row * (long long)LK;

    constexpr int CNT = LK / 32;
    float vals[CNT];
    float mx = -3.0e38f;
#pragma unroll
    for (int i = 0; i < CNT; ++i) { vals[i] = p[lane + (i << 5)]; mx = fmaxf(mx, vals[i]); }
#pragma unroll
    for (int off = 16; off > 0; off >>= 1) mx = fmaxf(mx, __shfl_xor(mx, off, 32));
    float s = 0.0f;
#pragma unroll
    for (int i = 0; i < CNT; ++i) { vals[i] = __expf(vals[i] - mx); s += vals[i]; }
#pragma unroll
    for (int off = 16; off > 0; off >>= 1) s += __shfl_xor(s, off, 32);
    const float inv = 1.0f / s;
#pragma unroll
    for (int i = 0; i < CNT; ++i) p[lane + (i << 5)] = vals[i] * inv;
}

// LayerNorm over rows of 1024 (bf16 in, fp32 out), one block per row.
__global__ __launch_bounds__(256)
void layernorm_bf16_kernel(const __bf16* __restrict__ x,
                           const float* __restrict__ g,
                           const float* __restrict__ b,
                           float* __restrict__ out)
{
    const long long row = blockIdx.x;
    const int tid  = threadIdx.x;
    const int lane = tid & 31;
    const int wid  = tid >> 5;

    const v4bf xv = *(const v4bf*)(x + row * 1024 + tid * 4);
    const float f0 = (float)xv[0], f1 = (float)xv[1], f2 = (float)xv[2], f3 = (float)xv[3];
    float s  = f0 + f1 + f2 + f3;
    float s2 = f0 * f0 + f1 * f1 + f2 * f2 + f3 * f3;
#pragma unroll
    for (int off = 16; off > 0; off >>= 1) {
        s  += __shfl_xor(s,  off, 32);
        s2 += __shfl_xor(s2, off, 32);
    }
    __shared__ float redS[8], redS2[8];
    if (lane == 0) { redS[wid] = s; redS2[wid] = s2; }
    __syncthreads();
    float ts = 0.0f, t2 = 0.0f;
#pragma unroll
    for (int i = 0; i < 8; ++i) { ts += redS[i]; t2 += redS2[i]; }
    const float mu   = ts * (1.0f / 1024.0f);
    const float var  = t2 * (1.0f / 1024.0f) - mu * mu;
    const float rstd = rsqrtf(var + 1e-5f);

    const float4 gv = *(const float4*)(g + tid * 4);
    const float4 bv = *(const float4*)(b + tid * 4);
    float4 o;
    o.x = (f0 - mu) * rstd * gv.x + bv.x;
    o.y = (f1 - mu) * rstd * gv.y + bv.y;
    o.z = (f2 - mu) * rstd * gv.z + bv.z;
    o.w = (f3 - mu) * rstd * gv.w + bv.w;
    *(float4*)(out + row * 1024 + tid * 4) = o;
}

// ---------------------------------------------------------------------------

extern "C" void kernel_launch(void* const* d_in, const int* in_sizes, int n_in,
                              void* d_out, int out_size, void* d_ws, size_t ws_size,
                              hipStream_t stream)
{
    (void)in_sizes; (void)n_in; (void)out_size; (void)ws_size;
    typedef long long ll;

    const int B = 16, H = 8, HID = 1024, HD = 128, Lq = 1024, Lk = 256;

    const float* lig    = (const float*)d_in[0];
    const float* aa     = (const float*)d_in[1];
    const int*   mask_l = (const int*)  d_in[2];
    const int*   mask_a = (const int*)  d_in[3];
    const float* Wsrc[8] = { (const float*)d_in[4],  (const float*)d_in[6],
                             (const float*)d_in[8],  (const float*)d_in[10],
                             (const float*)d_in[12], (const float*)d_in[14],
                             (const float*)d_in[16], (const float*)d_in[18] };
    const float* bq  = (const float*)d_in[5];  const float* bk  = (const float*)d_in[7];
    const float* bv  = (const float*)d_in[9];  const float* bo  = (const float*)d_in[11];
    const float* br1 = (const float*)d_in[13]; const float* br2 = (const float*)d_in[15];
    const float* bl1 = (const float*)d_in[17]; const float* bl2 = (const float*)d_in[19];
    const float* g_aa  = (const float*)d_in[20]; const float* b_aa  = (const float*)d_in[21];
    const float* g_lig = (const float*)d_in[22]; const float* b_lig = (const float*)d_in[23];

    // d_out: lig_out | aa_out | attn_aa_lig | attn_lig_aa
    float* out      = (float*)d_out;
    float* lig_out  = out;
    float* aa_out   = out + 4194304;
    float* attn_aa  = out + 4194304 + 16777216;              // [16,8,1024,256]
    float* attn_lig = attn_aa + 33554432;                    // [16,8,256,1024]

    // bf16 workspace with liveness-based reuse
    char* wsb = (char*)d_ws;
    auto alloc = [&](size_t bytes) { char* p = wsb; wsb += bytes; return p; };
    __bf16* aa_bf  = (__bf16*)alloc(33554432);   // [16*1024,1024]
    __bf16* lig_bf = (__bf16*)alloc(8388608);    // [16*256,1024]
    __bf16* Wt[8];
    for (int i = 0; i < 8; ++i) Wt[i] = (__bf16*)alloc(2097152);  // [N,K] bf16
    __bf16* Q_aa   = (__bf16*)alloc(33554432);
    __bf16* K_aa   = (__bf16*)alloc(33554432);
    __bf16* V_aa   = (__bf16*)alloc(33554432);
    __bf16* Q_lig  = (__bf16*)alloc(8388608);
    __bf16* K_lig  = (__bf16*)alloc(8388608);
    __bf16* V_lig  = (__bf16*)alloc(8388608);
    __bf16* ctx_aa  = (__bf16*)alloc(33554432);
    __bf16* ctx_lig = (__bf16*)alloc(8388608);
    __bf16* attO_aa  = Q_aa;    // Q dead after scores
    __bf16* attO_lig = Q_lig;
    __bf16* h_aa     = K_aa;    // K dead after scores
    __bf16* h_lig    = K_lig;
    __bf16* m_aa     = V_aa;    // V dead after ctx
    __bf16* m_lig    = V_lig;

    const dim3 blk(256);

    // ---- one-time casts ----
    cast_f32_bf16_kernel<<<dim3(16384), blk, 0, stream>>>(aa,  aa_bf,  4194304);
    cast_f32_bf16_kernel<<<dim3(4096),  blk, 0, stream>>>(lig, lig_bf, 1048576);
    for (int i = 0; i < 8; ++i)
        transpose_cast_kernel<<<dim3(32, 32), blk, 0, stream>>>(Wsrc[i], Wt[i]);

    // ---- GEMM wrappers ----
    auto gemmP = [&](const __bf16* A, const __bf16* Bw, __bf16* Cp, const float* bias,
                     const float* resid, int M, int leaky) {
        dim3 grid(HID / TILE_N, M / TILE_M, 1);
        gemm_wmma_kernel<1, 0, 1><<<grid, blk, 0, stream>>>(
            A, HID, 0, 0, Bw, HID, 0, 0, Cp, HID, 0, 0,
            bias, resid, HID, nullptr, 0, HID, 1, 1.0f, leaky);
    };
    const float sc = 0.088388347648318447f;   // 1/sqrt(128)
    auto gemmS = [&](const __bf16* Q, ll sQb, const __bf16* Kt, ll sKb,
                     float* Cp, int ldcS, ll sCbv, ll sChv,
                     const int* msk, int mstride, int M, int N) {
        dim3 grid(N / TILE_N, M / TILE_M, B * H);
        gemm_wmma_kernel<1, 0, 0><<<grid, blk, 0, stream>>>(
            Q, HID, sQb, HD, Kt, HID, sKb, HD, Cp, ldcS, sCbv, sChv,
            nullptr, nullptr, 0, msk, mstride, HD, H, sc, 0);
    };
    auto gemmC = [&](const float* attn, int ldaA, ll sAbv, ll sAhv,
                     const __bf16* V, ll sVb, __bf16* Cp, ll sCbv, int M, int Kd) {
        dim3 grid(HD / TILE_N, M / TILE_M, B * H);
        gemm_wmma_kernel<0, 1, 1><<<grid, blk, 0, stream>>>(
            attn, ldaA, sAbv, sAhv, V, HID, sVb, HD, Cp, HID, sCbv, HD,
            nullptr, nullptr, 0, nullptr, 0, Kd, H, 1.0f, 0);
    };

    // ---- projections ----
    gemmP(aa_bf,  Wt[0], Q_aa,  bq, nullptr, B * Lq, 0);
    gemmP(aa_bf,  Wt[1], K_aa,  bk, nullptr, B * Lq, 0);
    gemmP(aa_bf,  Wt[2], V_aa,  bv, nullptr, B * Lq, 0);
    gemmP(lig_bf, Wt[0], Q_lig, bq, nullptr, B * Lk, 0);
    gemmP(lig_bf, Wt[1], K_lig, bk, nullptr, B * Lk, 0);
    gemmP(lig_bf, Wt[2], V_lig, bv, nullptr, B * Lk, 0);

    // ---- aa queries over ligand keys ----
    gemmS(Q_aa, (ll)Lq * HID, K_lig, (ll)Lk * HID,
          attn_aa, Lk, (ll)H * Lq * Lk, (ll)Lq * Lk, mask_l, Lk, Lq, Lk);
    softmax_rows_kernel<256><<<dim3((B * H * Lq) / 8), blk, 0, stream>>>(attn_aa, (ll)B * H * Lq);
    gemmC(attn_aa, Lk, (ll)H * Lq * Lk, (ll)Lq * Lk,
          V_lig, (ll)Lk * HID, ctx_aa, (ll)Lq * HID, Lq, Lk);

    // ---- ligand queries over aa keys ----
    gemmS(Q_lig, (ll)Lk * HID, K_aa, (ll)Lq * HID,
          attn_lig, Lq, (ll)H * Lk * Lq, (ll)Lk * Lq, mask_a, Lq, Lk, Lq);
    softmax_rows_kernel<1024><<<dim3((B * H * Lk) / 8), blk, 0, stream>>>(attn_lig, (ll)B * H * Lk);
    gemmC(attn_lig, Lq, (ll)H * Lk * Lq, (ll)Lk * Lq,
          V_aa, (ll)Lq * HID, ctx_lig, (ll)Lk * HID, Lk, Lq);

    // ---- output projections ----
    gemmP(ctx_aa,  Wt[3], attO_aa,  bo, nullptr, B * Lq, 0);
    gemmP(ctx_lig, Wt[3], attO_lig, bo, nullptr, B * Lk, 0);

    // ---- MLPs: leaky(xW1+b1) then leaky(hW2+b2) + residual ----
    gemmP(attO_aa,  Wt[4], h_aa,  br1, nullptr, B * Lq, 1);
    gemmP(h_aa,     Wt[5], m_aa,  br2, aa,      B * Lq, 1);
    gemmP(attO_lig, Wt[6], h_lig, bl1, nullptr, B * Lk, 1);
    gemmP(h_lig,    Wt[7], m_lig, bl2, lig,     B * Lk, 1);

    // ---- LayerNorms -> final outputs ----
    layernorm_bf16_kernel<<<dim3(B * Lq), blk, 0, stream>>>(m_aa,  g_aa,  b_aa,  aa_out);
    layernorm_bf16_kernel<<<dim3(B * Lk), blk, 0, stream>>>(m_lig, g_lig, b_lig, lig_out);
}
